// MHA_16260746182713
// MI455X (gfx1250) — matrix operand order repaired and checked by reading, not compiled, not run on hardware
//
#include <hip/hip_runtime.h>
#include <hip/hip_bf16.h>

#define D_MODEL 1024
#define NUM_HEADS 16
#define DK 64
#define SEQ 2048
#define BATCH 4

typedef __attribute__((ext_vector_type(16))) __bf16 v16bf;
typedef __attribute__((ext_vector_type(8)))  float  v8f;

// f32 -> bf16 via the hardware convert (compiler lowers the cast).
__device__ __forceinline__ unsigned short f2bf(float f) {
  union { __bf16 b; unsigned short s; } u;
  u.b = (__bf16)f;
  return u.s;
}

// Load a 16-element bf16 fragment as two 16-byte chunks.
__device__ __forceinline__ v16bf load_bf_2x16B(const unsigned short* p0,
                                               const unsigned short* p1) {
  union { v16bf v; int4 q[2]; } u;
  u.q[0] = *(const int4*)p0;
  u.q[1] = *(const int4*)p1;
  return u.v;
}

// Build a bf16 fragment from two 8-float chunks (f32 source, hw cvt).
__device__ __forceinline__ v16bf cvt_bf_from_f32(const float* p0,
                                                 const float* p1) {
  v16bf v;
#pragma unroll
  for (int i = 0; i < 8; ++i) {
    v[i]     = (__bf16)p0[i];
    v[8 + i] = (__bf16)p1[i];
  }
  return v;
}

__device__ __forceinline__ v8f wmma_bf16(v16bf a, v16bf b, v8f c) {
  return __builtin_amdgcn_wmma_f32_16x16x32_bf16(
      /*neg_a=*/false, a, /*neg_b=*/false, b,
      /*c_mod=*/(short)0, c, /*reuse_a=*/false, /*reuse_b=*/false);
}

// XOR lane swizzle within the wave (group-of-32 mode, and=0x1f, xor=MASK).
template <int MASK>
__device__ __forceinline__ float swz_xor(float v) {
  return __int_as_float(
      __builtin_amdgcn_ds_swizzle(__float_as_int(v), (MASK << 10) | 0x1F));
}

// ---------------------------------------------------------------------------
// Kernel 0: bulk f32 -> bf16 conversion (memory-bound, ~5us of the budget).
// 8 elements per thread; element count must be a multiple of 2048.
// ---------------------------------------------------------------------------
__global__ __launch_bounds__(256) void cvt_bf16_kernel(
    const float* __restrict__ in, unsigned short* __restrict__ out) {
  const size_t i = ((size_t)blockIdx.x * 256 + threadIdx.x) * 8;
  const float4 a = *(const float4*)(in + i);
  const float4 b = *(const float4*)(in + i + 4);
  union { unsigned short s[8]; int4 q; } u;
  u.s[0] = f2bf(a.x); u.s[1] = f2bf(a.y); u.s[2] = f2bf(a.z); u.s[3] = f2bf(a.w);
  u.s[4] = f2bf(b.x); u.s[5] = f2bf(b.y); u.s[6] = f2bf(b.z); u.s[7] = f2bf(b.w);
  *(int4*)(out + i) = u.q;
}

// ---------------------------------------------------------------------------
// Double-buffered 16x64-per-wave bf16 GEMM core: C[16x64] += A[16xK] * B^T.
// A-fragment loaded once per K-step feeds 4 WMMAs; next K-step's 10 b128
// loads are issued before the current 4 WMMAs retire (software pipeline).
// ---------------------------------------------------------------------------
__device__ __forceinline__ void gemm_16x64(
    const unsigned short* __restrict__ arow,
    const unsigned short* __restrict__ br0,
    const unsigned short* __restrict__ br1,
    const unsigned short* __restrict__ br2,
    const unsigned short* __restrict__ br3,
    int c0, int k16, v8f& acc0, v8f& acc1, v8f& acc2, v8f& acc3) {
  v16bf a  = load_bf_2x16B(arow + 2 * c0, arow + 2 * c0 + 8);
  v16bf b0 = load_bf_2x16B(br0 + k16, br0 + k16 + 8);
  v16bf b1 = load_bf_2x16B(br1 + k16, br1 + k16 + 8);
  v16bf b2 = load_bf_2x16B(br2 + k16, br2 + k16 + 8);
  v16bf b3 = load_bf_2x16B(br3 + k16, br3 + k16 + 8);
  for (int kb = 32; kb < D_MODEL; kb += 32) {
    const v16bf an  = load_bf_2x16B(arow + kb + 2 * c0, arow + kb + 2 * c0 + 8);
    const v16bf bn0 = load_bf_2x16B(br0 + kb + k16, br0 + kb + k16 + 8);
    const v16bf bn1 = load_bf_2x16B(br1 + kb + k16, br1 + kb + k16 + 8);
    const v16bf bn2 = load_bf_2x16B(br2 + kb + k16, br2 + kb + k16 + 8);
    const v16bf bn3 = load_bf_2x16B(br3 + kb + k16, br3 + kb + k16 + 8);
    acc0 = wmma_bf16(a, b0, acc0);
    acc1 = wmma_bf16(a, b1, acc1);
    acc2 = wmma_bf16(a, b2, acc2);
    acc3 = wmma_bf16(a, b3, acc3);
    a = an; b0 = bn0; b1 = bn1; b2 = bn2; b3 = bn3;
  }
  acc0 = wmma_bf16(a, b0, acc0);
  acc1 = wmma_bf16(a, b1, acc1);
  acc2 = wmma_bf16(a, b2, acc2);
  acc3 = wmma_bf16(a, b3, acc3);
}

// ---------------------------------------------------------------------------
// Kernel 1: fused QKV projection (+ RoPE on Q/K), all-bf16 operands.
// y[row, e] = sum_d x[row, d] * w[e, d]   (row = b*S + s, e = h*64 + dd)
// One wave computes a 16x64 tile (exactly one head). blockIdx.y selects Q/K/V.
// Q/K stored bf16 [bh, S, 64] after RoPE; V stored bf16 transposed [bh, 64, S].
// ---------------------------------------------------------------------------
__global__ __launch_bounds__(256) void qkv_rope_kernel(
    const unsigned short* __restrict__ xb, const unsigned short* __restrict__ wqb,
    const unsigned short* __restrict__ wkb, const unsigned short* __restrict__ wvb,
    const int* __restrict__ pos,
    unsigned short* __restrict__ Q, unsigned short* __restrict__ K,
    unsigned short* __restrict__ Vt) {
  const int lane  = threadIdx.x & 31;
  const int wvid  = threadIdx.x >> 5;
  const int n16   = lane & 15;
  const int c0    = (lane < 16) ? 0 : 8;    // A-frag K chunk base
  const int k16   = (lane < 16) ? 0 : 16;   // B-frag K base
  const int mhalf = (lane >= 16) ? 8 : 0;   // C-frag row half

  const int t     = blockIdx.x * 8 + wvid;  // 512 m-tiles * 16 n64-tiles
  const int mbase = (t >> 4) * 16;          // token row tile
  const int nbase = (t & 15) * 64;          // feature tile (one head)
  const int sel   = blockIdx.y;
  const unsigned short* w = (sel == 0) ? wqb : (sel == 1) ? wkb : wvb;

  const unsigned short* arow = xb + (size_t)(mbase + n16) * D_MODEL;
  const unsigned short* br0 = w + (size_t)(nbase + 0  + n16) * D_MODEL;
  const unsigned short* br1 = w + (size_t)(nbase + 16 + n16) * D_MODEL;
  const unsigned short* br2 = w + (size_t)(nbase + 32 + n16) * D_MODEL;
  const unsigned short* br3 = w + (size_t)(nbase + 48 + n16) * D_MODEL;

  v8f acc0 = {0.f, 0.f, 0.f, 0.f, 0.f, 0.f, 0.f, 0.f};
  v8f acc1 = acc0, acc2 = acc0, acc3 = acc0;
  gemm_16x64(arow, br0, br1, br2, br3, c0, k16, acc0, acc1, acc2, acc3);

  const int h = nbase >> 6;  // one head per 64-wide tile
#pragma unroll
  for (int r = 0; r < 8; ++r) {
    const int m    = r + mhalf;
    const int row  = mbase + m;
    const int s    = row & (SEQ - 1);
    const int bidx = row >> 11;             // / SEQ
    const int bh   = bidx * NUM_HEADS + h;
    float vj[4] = {acc0[r], acc1[r], acc2[r], acc3[r]};
    if (sel == 2) {
      // V transposed: [bh, dd, s]
#pragma unroll
      for (int j = 0; j < 4; ++j) {
        const int dd = j * 16 + n16;
        Vt[((size_t)bh * DK + dd) * SEQ + s] = f2bf(vj[j]);
      }
    } else {
      unsigned short* dst = (sel == 0) ? Q : K;
      const float fpos = (float)pos[s];
#pragma unroll
      for (int j = 0; j < 4; ++j) {
        const int dd = j * 16 + n16;
        const float v = vj[j];
        // RoPE: pair partner sits in the adjacent lane (n and n^1).
        const float p   = swz_xor<1>(v);
        const int dd0   = dd & ~1;
        // theta^(-dd0/64) = exp(-dd0/64 * ln(10000))
        const float fr  = __expf(-(float)dd0 * (9.210340371976184f / 64.f));
        const float ang = fpos * fr;
        const float sn = __sinf(ang), cs = __cosf(ang);
        const float o = (dd & 1) ? (p * sn + v * cs) : (v * cs - p * sn);
        dst[((size_t)bh * SEQ + s) * DK + dd] = f2bf(o);
      }
    }
  }
}

// ---------------------------------------------------------------------------
// Kernel 2: causal flash attention. One wave owns a 16-query tile; online
// softmax over 32-key blocks. All 8 fragment loads (4 K, 4 V) are hoisted to
// the top of each key-block iteration so the V loads hide under the softmax.
// P re-layout (C-frag f32 -> A-frag bf16) goes through a wave-private LDS
// patch with an explicit s_wait_dscnt.
// Output O: bf16 [b, s, h*64+dd] = [8192, 1024].
// ---------------------------------------------------------------------------
__global__ __launch_bounds__(256) void attn_kernel(
    const unsigned short* __restrict__ Q, const unsigned short* __restrict__ K,
    const unsigned short* __restrict__ Vt, unsigned short* __restrict__ O) {
  __shared__ float lds[8][16 * 32];  // per-wave 16x32 P patch (16 KB total)

  const int lane  = threadIdx.x & 31;
  const int wvid  = threadIdx.x >> 5;
  const int n16   = lane & 15;
  const int c0    = (lane < 16) ? 0 : 8;
  const int k16   = (lane < 16) ? 0 : 16;
  const int mhalf = (lane >= 16) ? 8 : 0;

  const int bh    = blockIdx.x >> 4;        // 16 q-chunks per (b,h)
  const int qbase = ((blockIdx.x & 15) * 8 + wvid) * 16;
  const int b     = bh >> 4;
  const int h     = bh & 15;

  const unsigned short* Qb = Q + (size_t)bh * SEQ * DK;
  const unsigned short* Kb = K + (size_t)bh * SEQ * DK;
  const unsigned short* Vb = Vt + (size_t)bh * DK * SEQ;

  // Q A-fragments (d = 0..31 and 32..63), loaded once.
  const unsigned short* qrow = Qb + (size_t)(qbase + n16) * DK;
  const v16bf qa0 = load_bf_2x16B(qrow + 2 * c0, qrow + 2 * c0 + 8);
  const v16bf qa1 = load_bf_2x16B(qrow + 32 + 2 * c0, qrow + 40 + 2 * c0);

  v8f a0 = {0.f,0.f,0.f,0.f,0.f,0.f,0.f,0.f};
  v8f a1 = a0, a2 = a0, a3 = a0;
  float mst[8], lst[8];
#pragma unroll
  for (int r = 0; r < 8; ++r) { mst[r] = -1e30f; lst[r] = 0.f; }

  float* lp = &lds[wvid][0];
  const int nkb = (qbase + 15) >> 5;  // last 32-key block touching the diagonal
  for (int kb = 0; kb <= nkb; ++kb) {
    const int kb0 = kb * 32;

    // --- hoisted fragment loads: 4 K B-frags + 4 V B-frags -------------------
    const unsigned short* kpa = Kb + (size_t)(kb0 + n16) * DK + k16;
    const unsigned short* kpb = Kb + (size_t)(kb0 + 16 + n16) * DK + k16;
    const v16bf kf00 = load_bf_2x16B(kpa, kpa + 8);        // keys 0..15, d 0..31
    const v16bf kf01 = load_bf_2x16B(kpa + 32, kpa + 40);  // keys 0..15, d 32..63
    const v16bf kf10 = load_bf_2x16B(kpb, kpb + 8);        // keys 16..31, d 0..31
    const v16bf kf11 = load_bf_2x16B(kpb + 32, kpb + 40);  // keys 16..31, d 32..63
    const unsigned short* vp0 = Vb + (size_t)(0  + n16) * SEQ + kb0 + k16;
    const unsigned short* vp1 = Vb + (size_t)(16 + n16) * SEQ + kb0 + k16;
    const unsigned short* vp2 = Vb + (size_t)(32 + n16) * SEQ + kb0 + k16;
    const unsigned short* vp3 = Vb + (size_t)(48 + n16) * SEQ + kb0 + k16;
    const v16bf vf0 = load_bf_2x16B(vp0, vp0 + 8);
    const v16bf vf1 = load_bf_2x16B(vp1, vp1 + 8);
    const v16bf vf2 = load_bf_2x16B(vp2, vp2 + 8);
    const v16bf vf3 = load_bf_2x16B(vp3, vp3 + 8);

    // --- scores: two 16x16 tiles over 32 keys --------------------------------
    v8f s0 = {0.f,0.f,0.f,0.f,0.f,0.f,0.f,0.f};
    v8f s1 = s0;
    s0 = wmma_bf16(qa0, kf00, s0);
    s0 = wmma_bf16(qa1, kf01, s0);
    s1 = wmma_bf16(qa0, kf10, s1);
    s1 = wmma_bf16(qa1, kf11, s1);

    // --- online softmax (row stats live per C-register; 16-lane reductions) --
    const float scale = 0.125f;  // 1/sqrt(64)
#pragma unroll
    for (int r = 0; r < 8; ++r) {
      const int m = r + mhalf;
      const int qrowi = qbase + m;
      float v0 = s0[r] * scale;
      float v1 = s1[r] * scale;
      v0 = (kb0 + n16 > qrowi) ? -1e30f : v0;
      v1 = (kb0 + 16 + n16 > qrowi) ? -1e30f : v1;
      float mx = fmaxf(v0, v1);
      mx = fmaxf(mx, swz_xor<1>(mx));
      mx = fmaxf(mx, swz_xor<2>(mx));
      mx = fmaxf(mx, swz_xor<4>(mx));
      mx = fmaxf(mx, swz_xor<8>(mx));
      const float mnew = fmaxf(mst[r], mx);
      const float corr = __expf(mst[r] - mnew);
      mst[r] = mnew;
      const float e0 = __expf(v0 - mnew);
      const float e1 = __expf(v1 - mnew);
      float rs = e0 + e1;
      rs += swz_xor<1>(rs);
      rs += swz_xor<2>(rs);
      rs += swz_xor<4>(rs);
      rs += swz_xor<8>(rs);
      lst[r] = lst[r] * corr + rs;
      a0[r] *= corr; a1[r] *= corr; a2[r] *= corr; a3[r] *= corr;
      lp[m * 32 + n16]      = e0;   // P row-major 16x32 in LDS
      lp[m * 32 + 16 + n16] = e1;
    }
    asm volatile("s_wait_dscnt 0x0" ::: "memory");

    // --- P (16x32 f32 in LDS) -> bf16 A-fragment (hw cvt) --------------------
    const float* prow = lp + n16 * 32;
    const v16bf pa = cvt_bf_from_f32(prow + 2 * c0, prow + 2 * c0 + 8);

    // --- PV: out[q, d] += P * V (V fragments already resident) ---------------
    a0 = wmma_bf16(pa, vf0, a0);
    a1 = wmma_bf16(pa, vf1, a1);
    a2 = wmma_bf16(pa, vf2, a2);
    a3 = wmma_bf16(pa, vf3, a3);
  }

  // --- normalize and store bf16 [b*S + s, h*64 + d] --------------------------
#pragma unroll
  for (int r = 0; r < 8; ++r) {
    const int m = r + mhalf;
    const int srow = qbase + m;
    const float inv = 1.f / lst[r];
    const size_t base = (size_t)(b * SEQ + srow) * D_MODEL + h * DK;
    O[base + 0  + n16] = f2bf(a0[r] * inv);
    O[base + 16 + n16] = f2bf(a1[r] * inv);
    O[base + 32 + n16] = f2bf(a2[r] * inv);
    O[base + 48 + n16] = f2bf(a3[r] * inv);
  }
}

// ---------------------------------------------------------------------------
// Kernel 3: output projection. out[row, d] = sum_e O[row, e] * wo[d, e].
// 16x64 tile per wave, all-bf16 operands, f32 result.
// ---------------------------------------------------------------------------
__global__ __launch_bounds__(256) void oproj_kernel(
    const unsigned short* __restrict__ O, const unsigned short* __restrict__ wob,
    float* __restrict__ out) {
  const int lane  = threadIdx.x & 31;
  const int wvid  = threadIdx.x >> 5;
  const int n16   = lane & 15;
  const int c0    = (lane < 16) ? 0 : 8;
  const int k16   = (lane < 16) ? 0 : 16;
  const int mhalf = (lane >= 16) ? 8 : 0;

  const int t     = blockIdx.x * 8 + wvid;
  const int mbase = (t >> 4) * 16;
  const int nbase = (t & 15) * 64;

  const unsigned short* arow = O + (size_t)(mbase + n16) * D_MODEL;
  const unsigned short* br0 = wob + (size_t)(nbase + 0  + n16) * D_MODEL;
  const unsigned short* br1 = wob + (size_t)(nbase + 16 + n16) * D_MODEL;
  const unsigned short* br2 = wob + (size_t)(nbase + 32 + n16) * D_MODEL;
  const unsigned short* br3 = wob + (size_t)(nbase + 48 + n16) * D_MODEL;

  v8f acc0 = {0.f, 0.f, 0.f, 0.f, 0.f, 0.f, 0.f, 0.f};
  v8f acc1 = acc0, acc2 = acc0, acc3 = acc0;
  gemm_16x64(arow, br0, br1, br2, br3, c0, k16, acc0, acc1, acc2, acc3);

#pragma unroll
  for (int r = 0; r < 8; ++r) {
    float* orow = out + (size_t)(mbase + r + mhalf) * D_MODEL + nbase;
    orow[0  + n16] = acc0[r];
    orow[16 + n16] = acc1[r];
    orow[32 + n16] = acc2[r];
    orow[48 + n16] = acc3[r];
  }
}

// ---------------------------------------------------------------------------
extern "C" void kernel_launch(void* const* d_in, const int* in_sizes, int n_in,
                              void* d_out, int out_size, void* d_ws,
                              size_t ws_size, hipStream_t stream) {
  (void)in_sizes; (void)n_in; (void)out_size; (void)ws_size;
  const float* x   = (const float*)d_in[0];
  const float* wq  = (const float*)d_in[1];
  const float* wk  = (const float*)d_in[2];
  const float* wvm = (const float*)d_in[3];
  const float* wo  = (const float*)d_in[4];
  const int*   pos = (const int*)d_in[5];

  // Workspace layout (bf16 ushort elements):
  //   xb  : B*S*D        = 8Mi  (16 MB)
  //   wqb/wkb/wvb/wob    : 1Mi each (2 MB each)
  //   Q, K (RoPE'd [bh,S,64]), Vt ([bh,64,S]), O ([B*S, D]) : 8Mi each
  const size_t n  = (size_t)BATCH * SEQ * D_MODEL;  // 8Mi
  const size_t wn = (size_t)D_MODEL * D_MODEL;      // 1Mi
  unsigned short* xb  = (unsigned short*)d_ws;
  unsigned short* wqb = xb + n;
  unsigned short* wkb = wqb + wn;
  unsigned short* wvb = wkb + wn;
  unsigned short* wob = wvb + wn;
  unsigned short* Q   = wob + wn;
  unsigned short* K   = Q + n;
  unsigned short* Vt  = K + n;
  unsigned short* O   = Vt + n;
  float* out = (float*)d_out;

  dim3 blk(256, 1, 1);
  // Bulk conversions: 2048 elements per block.
  cvt_bf16_kernel<<<dim3((unsigned)(n / 2048)), blk, 0, stream>>>(x, xb);
  cvt_bf16_kernel<<<dim3((unsigned)(wn / 2048)), blk, 0, stream>>>(wq, wqb);
  cvt_bf16_kernel<<<dim3((unsigned)(wn / 2048)), blk, 0, stream>>>(wk, wkb);
  cvt_bf16_kernel<<<dim3((unsigned)(wn / 2048)), blk, 0, stream>>>(wvm, wvb);
  cvt_bf16_kernel<<<dim3((unsigned)(wn / 2048)), blk, 0, stream>>>(wo, wob);

  // 512 m-tiles * 16 n64-tiles = 8192 wave-tiles -> 1024 blocks; y selects Q/K/V.
  qkv_rope_kernel<<<dim3(1024, 3, 1), blk, 0, stream>>>(xb, wqb, wkb, wvb, pos,
                                                        Q, K, Vt);
  // 64 (b,h) * 16 q-chunks = 1024 blocks, 8 query tiles (waves) each.
  attn_kernel<<<dim3(1024, 1, 1), blk, 0, stream>>>(Q, K, Vt, O);
  // Output projection GEMM.
  oproj_kernel<<<dim3(1024, 1, 1), blk, 0, stream>>>(O, wob, out);
}
